// MoEMerged_18219251269841
// MI455X (gfx1250) — matrix operand order repaired and checked by reading, not compiled, not run on hardware
//
#include <hip/hip_runtime.h>

// ---- problem dims (fixed by the reference) ----
#define S_SITES   1024
#define A_ALLELES 2048
#define RPA0      20
#define RPA1      10
#define R0_READS  40960   // A * RPA0
#define R1_READS  20480   // A * RPA1
#define CIN       10
#define LLEN      100
#define FDIM      32
#define KW        5       // conv kernel width; im2col K = CIN*KW = 50, padded to 64
#define HDIM      64
#define LOUT      96      // L - K + 1
#define CPAD      13      // channels incl. zero-pad so K/5 <= 12 is always in-bounds
#define TSTRIDE   104     // padded t stride (halves)

typedef __attribute__((ext_vector_type(16))) _Float16 v16h;
typedef __attribute__((ext_vector_type(8)))  float    v8f;

// ---------------------------------------------------------------------------
// Kernel 1: conv1d + bias + relu + mean over t  ->  feat[r, 32]
// One wave32 per read. im2col GEMM: W[32 x 50pad64] x X'[50pad64 x 96]
// via v_wmma_f32_16x16x32_f16: 2 mtiles x 6 ntiles x 2 ksteps = 24 WMMA/read.
// Zero-padding lives in LDS (channels 10..12), so the fragment gather is
// branch-free: plain ds_load_u16 with lane-constant offsets.
// __launch_bounds__(256, 4): cap at 256 VGPRs (no s_set_vgpr_msb churn),
// >=4 waves/SIMD for latency hiding; unroll 2 keeps only 4 B fragments live.
// ---------------------------------------------------------------------------
__global__ __launch_bounds__(256, 4) void conv_feat_wmma(
    const float* __restrict__ x,     // [R, CIN, LLEN]
    const float* __restrict__ cw,    // [FDIM, CIN, KW]
    const float* __restrict__ cb,    // [FDIM]
    float* __restrict__ feat,        // [R, FDIM]
    int R)
{
  // A fragments (weights) in the documented 16-bit A 16x32 lane layout,
  // pre-baked so each lane reads its 16 halves contiguously.
  __shared__ _Float16 sA[2][2][32][16];              // [mtile][kstep][lane][half]
  __shared__ _Float16 sX[8][CPAD][TSTRIDE];          // per-wave read slab (padded)

  const int tid  = threadIdx.x;
  const int lane = tid & 31;
  const int wv   = tid >> 5;
  const int hi16 = (lane >> 4) & 1;                  // 0: lanes 0-15, 1: lanes 16-31

  // Build A fragments cooperatively (zero-pad K=50..63).
  for (int i = tid; i < 2 * 2 * 32 * 16; i += 256) {
    int hh = i & 15;
    int ln = (i >> 4) & 31;
    int ks = (i >> 9) & 1;
    int mt = (i >> 10) & 1;
    int vg = hh >> 1, p = hh & 1;
    // A 16x32 f16 layout: lanes 0-15 get K 0-7 (V0-3) & 16-23 (V4-7);
    // lanes 16-31 get K 8-15 & 24-31.
    int K = ks * 32 + ((vg < 4) ? 0 : 16) + ((ln < 16) ? 0 : 8) + ((vg & 3) << 1) + p;
    int M = mt * 16 + (ln & 15);
    sA[mt][ks][ln][hh] =
        (K < CIN * KW) ? (_Float16)cw[M * (CIN * KW) + K] : (_Float16)0.0f;
  }
  __syncthreads();

  v16h afrag[2][2];
  #pragma unroll
  for (int mt = 0; mt < 2; ++mt)
    #pragma unroll
    for (int ks = 0; ks < 2; ++ks)
      #pragma unroll
      for (int hh = 0; hh < 16; ++hh)
        afrag[mt][ks][hh] = sA[mt][ks][lane][hh];

  // bias per (mt, j) for this lane's C-fragment rows: M = mt*16 + j + 8*hi16
  float breg[2][8];
  #pragma unroll
  for (int mt = 0; mt < 2; ++mt)
    #pragma unroll
    for (int j = 0; j < 8; ++j)
      breg[mt][j] = cb[mt * 16 + j + hi16 * 8];

  // Zero the padding channels (10..12) of this wave's slab once; the read
  // loop only overwrites channels 0..9, so pad stays zero across iterations.
  _Float16* sxw = &sX[wv][0][0];
  for (int i = lane; i < (CPAD - CIN) * TSTRIDE; i += 32)
    sxw[CIN * TSTRIDE + i] = (_Float16)0.0f;

  // Lane-constant im2col gather offsets for the B fragments.
  // B 32x16 f16 layout (matches A): lanes 0-15 hold K 0-7 & 16-23,
  // lanes 16-31 hold K 8-15 & 24-31; N = lane % 16.
  const int koffB = hi16 * 8;
  int ofs0[16], ofs1[16];
  #pragma unroll
  for (int hh = 0; hh < 16; ++hh) {
    int K0 = koffB + hh + ((hh >> 3) << 3); // kstep 0: K in [0,32)
    ofs0[hh] = (K0 / KW) * TSTRIDE + (K0 % KW);
    int K1 = 32 + koffB + hh + ((hh >> 3) << 3); // kstep 1: K in [32,64), pad in LDS
    ofs1[hh] = (K1 / KW) * TSTRIDE + (K1 % KW);
  }

  const int nwaves = (gridDim.x * blockDim.x) >> 5;
  const int gwave  = (blockIdx.x * 256 + tid) >> 5;
  const int tb     = lane & 15;            // N = lane % 16

  for (int r = gwave; r < R; r += nwaves) {
    const float* xr = x + (size_t)r * (CIN * LLEN);
    if (r + nwaves < R)
      __builtin_prefetch(x + (size_t)(r + nwaves) * (CIN * LLEN), 0, 0);

    // Stage read into LDS as f16: float4 global loads (b128, coalesced),
    // cvt in VGPRs, packed 8-byte LDS stores. 250 float4 groups; LLEN%4==0
    // so a group never crosses a channel; all addresses 8B-aligned.
    {
      const float4* xr4 = (const float4*)xr;
      for (int i = lane; i < (CIN * LLEN) / 4; i += 32) {
        float4 v4 = xr4[i];
        int c  = i / (LLEN / 4);
        int tt = (i % (LLEN / 4)) * 4;
        union { _Float16 h[4]; unsigned long long u; } pk;
        pk.h[0] = (_Float16)v4.x; pk.h[1] = (_Float16)v4.y;
        pk.h[2] = (_Float16)v4.z; pk.h[3] = (_Float16)v4.w;
        *(unsigned long long*)&sxw[c * TSTRIDE + tt] = pk.u;
      }
    }
    // per-wave LDS ops are in-order; compiler inserts s_wait_dscnt as needed

    float fsum[2][8] = {};                 // relu-sum over t, per C row

    #pragma unroll 2
    for (int nt = 0; nt < 6; ++nt) {
      const int tbase = nt * 16 + tb;
      v16h b0, b1;
      #pragma unroll
      for (int hh = 0; hh < 16; ++hh) {
        b0[hh] = sxw[ofs0[hh] + tbase];
        b1[hh] = sxw[ofs1[hh] + tbase];
      }
      #pragma unroll
      for (int mt = 0; mt < 2; ++mt) {
        v8f c = {};
        c = __builtin_amdgcn_wmma_f32_16x16x32_f16(false, afrag[mt][0],
                                                   false, b0, (short)0, c,
                                                   false, false);
        c = __builtin_amdgcn_wmma_f32_16x16x32_f16(false, afrag[mt][1],
                                                   false, b1, (short)0, c,
                                                   false, false);
        #pragma unroll
        for (int j = 0; j < 8; ++j)
          fsum[mt][j] += fmaxf(c[j] + breg[mt][j], 0.0f);
      }
    }

    // Reduce over N (t) across each 16-lane half; lanes 0 and 16 hold sums.
    #pragma unroll
    for (int mt = 0; mt < 2; ++mt)
      #pragma unroll
      for (int j = 0; j < 8; ++j) {
        float s = fsum[mt][j];
        s += __shfl_xor(s, 1, 32);
        s += __shfl_xor(s, 2, 32);
        s += __shfl_xor(s, 4, 32);
        s += __shfl_xor(s, 8, 32);
        if ((lane & 15) == 0)
          feat[(size_t)r * FDIM + mt * 16 + j + hi16 * 8] = s * (1.0f / 96.0f);
      }
  }
}

// ---------------------------------------------------------------------------
// Kernel 2: per-allele segment sum (segments are contiguous blocks of rpa).
// ---------------------------------------------------------------------------
__global__ __launch_bounds__(256) void allele_sum_kernel(
    const float* __restrict__ feat, float* __restrict__ al, int rpa)
{
  int idx = blockIdx.x * blockDim.x + threadIdx.x;   // a * 32 + f
  if (idx >= A_ALLELES * FDIM) return;
  int a = idx >> 5, f = idx & 31;
  const float* p = feat + (size_t)a * rpa * FDIM + f;
  float s = 0.0f;
  for (int i = 0; i < rpa; ++i) s += p[(size_t)i * FDIM];
  al[idx] = s;
}

// ---------------------------------------------------------------------------
// Kernel 3: meta head. site feats derive from allele sums (site = 2 alleles).
// ---------------------------------------------------------------------------
__global__ __launch_bounds__(256) void meta_kernel(
    const float* __restrict__ al0, const float* __restrict__ al1,
    const float* __restrict__ mw,  const float* __restrict__ mb,
    float* __restrict__ out)       // [S, 3]
{
  int s = blockIdx.x * blockDim.x + threadIdx.x;
  if (s >= S_SITES) return;
  float l0 = mb[0], l1 = mb[1], l2 = mb[2];
  for (int f = 0; f < FDIM; ++f) {
    float v = al0[(2 * s) * FDIM + f] + al0[(2 * s + 1) * FDIM + f];
    l0 += v * mw[f * 3 + 0]; l1 += v * mw[f * 3 + 1]; l2 += v * mw[f * 3 + 2];
  }
  for (int f = 0; f < FDIM; ++f) {
    float v = al1[(2 * s) * FDIM + f] + al1[(2 * s + 1) * FDIM + f];
    int j = FDIM + f;
    l0 += v * mw[j * 3 + 0]; l1 += v * mw[j * 3 + 1]; l2 += v * mw[j * 3 + 2];
  }
  float m  = fmaxf(l0, fmaxf(l1, l2));
  float e0 = __expf(l0 - m), e1 = __expf(l1 - m), e2 = __expf(l2 - m);
  float inv = 1.0f / (e0 + e1 + e2);
  out[s * 3 + 0] = e0 * inv;
  out[s * 3 + 1] = e1 * inv;
  out[s * 3 + 2] = e2 * inv;
}

// ---------------------------------------------------------------------------
// Kernel 4: expert MLP + per-site 2-way softmax. f1==nullptr -> Fin=32.
// ---------------------------------------------------------------------------
__global__ __launch_bounds__(256) void expert_kernel(
    const float* __restrict__ f0, const float* __restrict__ f1,
    const float* __restrict__ w1, const float* __restrict__ b1,
    const float* __restrict__ w2, const float* __restrict__ b2,
    float* __restrict__ out)       // [A]
{
  int s = blockIdx.x * blockDim.x + threadIdx.x;
  if (s >= S_SITES) return;
  float lg[2];
  for (int j = 0; j < 2; ++j) {
    int a = 2 * s + j;
    float logit = b2[0];
    for (int n = 0; n < HDIM; ++n) {
      float acc = b1[n];
      #pragma unroll 8
      for (int f = 0; f < FDIM; ++f)
        acc = fmaf(f0[(size_t)a * FDIM + f], w1[f * HDIM + n], acc);
      if (f1) {
        #pragma unroll 8
        for (int f = 0; f < FDIM; ++f)
          acc = fmaf(f1[(size_t)a * FDIM + f], w1[(FDIM + f) * HDIM + n], acc);
      }
      logit = fmaf(fmaxf(acc, 0.0f), w2[n], logit);
    }
    lg[j] = logit;
  }
  float m  = fmaxf(lg[0], lg[1]);
  float e0 = __expf(lg[0] - m), e1 = __expf(lg[1] - m);
  float inv = 1.0f / (e0 + e1);
  out[2 * s + 0] = e0 * inv;
  out[2 * s + 1] = e1 * inv;
}

// ---------------------------------------------------------------------------
extern "C" void kernel_launch(void* const* d_in, const int* in_sizes, int n_in,
                              void* d_out, int out_size, void* d_ws, size_t ws_size,
                              hipStream_t stream) {
  (void)in_sizes; (void)n_in; (void)out_size; (void)ws_size;

  const float* tensors0 = (const float*)d_in[0];
  const float* tensors1 = (const float*)d_in[1];
  const float* conv0_w  = (const float*)d_in[2];
  const float* conv0_b  = (const float*)d_in[3];
  const float* conv1_w  = (const float*)d_in[4];
  const float* conv1_b  = (const float*)d_in[5];
  const float* meta_w   = (const float*)d_in[6];
  const float* meta_b   = (const float*)d_in[7];
  const float* ngs_w1   = (const float*)d_in[8];
  const float* ngs_b1   = (const float*)d_in[9];
  const float* ngs_w2   = (const float*)d_in[10];
  const float* ngs_b2   = (const float*)d_in[11];
  const float* tgs_w1   = (const float*)d_in[12];
  const float* tgs_b1   = (const float*)d_in[13];
  const float* tgs_w2   = (const float*)d_in[14];
  const float* tgs_b2   = (const float*)d_in[15];
  const float* hyb_w1   = (const float*)d_in[16];
  const float* hyb_b1   = (const float*)d_in[17];
  const float* hyb_w2   = (const float*)d_in[18];
  const float* hyb_b2   = (const float*)d_in[19];
  // d_in[20..24]: segment id arrays — segments are contiguous fixed-size
  // blocks (repeat(arange)), so indices are implied by layout.

  float* ws    = (float*)d_ws;
  float* feat0 = ws;                                     // [R0, 32]
  float* feat1 = feat0 + (size_t)R0_READS * FDIM;        // [R1, 32]
  float* al0   = feat1 + (size_t)R1_READS * FDIM;        // [A, 32]
  float* al1   = al0 + (size_t)A_ALLELES * FDIM;         // [A, 32]

  float* out      = (float*)d_out;
  float* out_ngs  = out;                     // [2048]
  float* out_tgs  = out + A_ALLELES;         // [2048]
  float* out_hyb  = out + 2 * A_ALLELES;     // [2048]
  float* out_meta = out + 3 * A_ALLELES;     // [1024*3]

  // 8 waves/block, 1 read per wave.
  conv_feat_wmma<<<R0_READS / 8, 256, 0, stream>>>(tensors0, conv0_w, conv0_b,
                                                   feat0, R0_READS);
  conv_feat_wmma<<<R1_READS / 8, 256, 0, stream>>>(tensors1, conv1_w, conv1_b,
                                                   feat1, R1_READS);

  allele_sum_kernel<<<(A_ALLELES * FDIM) / 256, 256, 0, stream>>>(feat0, al0, RPA0);
  allele_sum_kernel<<<(A_ALLELES * FDIM) / 256, 256, 0, stream>>>(feat1, al1, RPA1);

  meta_kernel<<<S_SITES / 256, 256, 0, stream>>>(al0, al1, meta_w, meta_b, out_meta);

  expert_kernel<<<S_SITES / 256, 256, 0, stream>>>(al0, nullptr, ngs_w1, ngs_b1,
                                                   ngs_w2, ngs_b2, out_ngs);
  expert_kernel<<<S_SITES / 256, 256, 0, stream>>>(al1, nullptr, tgs_w1, tgs_b1,
                                                   tgs_w2, tgs_b2, out_tgs);
  expert_kernel<<<S_SITES / 256, 256, 0, stream>>>(al0, al1, hyb_w1, hyb_b1,
                                                   hyb_w2, hyb_b2, out_hyb);
}
